// SimGNN_49555332661649
// MI455X (gfx1250) — matrix operand order repaired and checked by reading, not compiled, not run on hardware
//
#include <hip/hip_runtime.h>
#include <math.h>

// ---------------------------------------------------------------------------
// SimGNN forward for MI455X (gfx1250, wave32).
// All heavy matmuls (GCN layers + 20000x20000x32 similarity scores) run on
// v_wmma_f32_16x16x32_bf16 with operands PRE-PACKED into the wave32 WMMA
// fragment layout (bf16, 32B per lane per fragment) so the inner loops are
// pure b128 loads + WMMA.  Score passes block 4 M-tiles per wave: one B
// fragment load feeds 4 WMMAs whose binning VALU co-executes with the
// matrix pipe.  Packed score operands are ~1.3MB each -> L2 resident.
// ---------------------------------------------------------------------------

typedef __attribute__((ext_vector_type(16))) __bf16 v16bf;
typedef __attribute__((ext_vector_type(8)))  float  v8f;

#define FRAG 512  // bf16 elements per packed 16x16x32 fragment (32 lanes x 16)
#define MT 4      // M-tiles per wave in the score passes

// ---- float -> bf16 (round to nearest even) --------------------------------
__device__ __forceinline__ __bf16 f2bf(float f) {
  unsigned u = __builtin_bit_cast(unsigned, f);
  unsigned r = u + 0x7FFFu + ((u >> 16) & 1u);
  unsigned short h = (unsigned short)(r >> 16);
  return __builtin_bit_cast(__bf16, h);
}

// ---- ordered float <-> uint keys for atomic min/max -----------------------
__device__ __forceinline__ unsigned fkey(float f) {
  unsigned u = __builtin_bit_cast(unsigned, f);
  return (u & 0x80000000u) ? ~u : (u | 0x80000000u);
}
__device__ __forceinline__ float fromkey(unsigned k) {
  unsigned u = (k & 0x80000000u) ? (k ^ 0x80000000u) : ~k;
  return __builtin_bit_cast(float, u);
}

// A-fragment K index for half i, lane-group g (contiguous: K = 16*g + i, matches B packing)
__device__ __forceinline__ int a_k_of(int i, int g) {
  return 16 * g + i;
}

// ---------------------------------------------------------------------------
// Fragment packing kernels: gather + f32->bf16 convert ONCE per tile.
// 8 waves/block, one fragment per wave; lane writes its 16 bf16 (32B) chunk.
// ---------------------------------------------------------------------------

// A fragments from row-major X[M x K]; frag index = tm * kchunks + kc.
__global__ void pack_a_frags(const float* __restrict__ X,
                             __bf16* __restrict__ PA, int mtiles, int K) {
  int kchunks = K >> 5;
  int frag = blockIdx.x * 8 + (threadIdx.x >> 5);
  if (frag >= mtiles * kchunks) return;
  int tm = frag / kchunks, kc = frag - tm * kchunks;
  int lane = threadIdx.x & 31;
  int m = lane & 15, g = (lane >> 4) & 1;
  const float* r = X + (size_t)(tm * 16 + m) * K + kc * 32;
  v16bf a;
#pragma unroll
  for (int i = 0; i < 16; ++i) a[i] = f2bf(r[a_k_of(i, g)]);
  *(v16bf*)(PA + (size_t)frag * FRAG + lane * 16) = a;
}

// B fragments from row-major W[K x Nout]; frag index = kc * ntiles + tn.
// B layout: column = lane&15, half j -> K = 16*(lane>>4) + j.
__global__ void pack_b_frags(const float* __restrict__ W,
                             __bf16* __restrict__ PW, int K, int Nout) {
  int ntiles = Nout >> 4, kchunks = K >> 5;
  int frag = blockIdx.x * 8 + (threadIdx.x >> 5);
  if (frag >= kchunks * ntiles) return;
  int kc = frag / ntiles, tn = frag - kc * ntiles;
  int lane = threadIdx.x & 31;
  int n = lane & 15, g = (lane >> 4) & 1;
  v16bf b;
#pragma unroll
  for (int j = 0; j < 16; ++j)
    b[j] = f2bf(W[(size_t)(kc * 32 + 16 * g + j) * Nout + tn * 16 + n]);
  *(v16bf*)(PW + (size_t)frag * FRAG + lane * 16) = b;
}

// B fragments of A2^T from row-major A2[N x 32] (K = 32, one chunk):
// B[k][n] = A2[tn*16+n][k] -> lane reads 16 consecutive floats of its row.
__global__ void pack_bt_frags(const float* __restrict__ A2,
                              __bf16* __restrict__ PB, int ntiles) {
  int frag = blockIdx.x * 8 + (threadIdx.x >> 5);
  if (frag >= ntiles) return;
  int lane = threadIdx.x & 31;
  int n = lane & 15, g = (lane >> 4) & 1;
  const float* r = A2 + (size_t)(frag * 16 + n) * 32 + 16 * g;
  v16bf b;
#pragma unroll
  for (int j = 0; j < 16; ++j) b[j] = f2bf(r[j]);
  *(v16bf*)(PB + (size_t)frag * FRAG + lane * 16) = b;
}

// ---------------------------------------------------------------------------
// Generic fill
// ---------------------------------------------------------------------------
__global__ void fill_u32(unsigned* __restrict__ p, unsigned v, long n) {
  long i = (long)blockIdx.x * blockDim.x + threadIdx.x;
  if (i < n) p[i] = v;
}

// ---------------------------------------------------------------------------
// GEMM on packed fragments: Y[M x Nout] = X @ W.  One wave per 16x16 tile.
// ---------------------------------------------------------------------------
__global__ void gemm_packed(const __bf16* __restrict__ PA,
                            const __bf16* __restrict__ PW,
                            float* __restrict__ Y, int mtiles, int kchunks,
                            int ntiles, int Nout) {
  int wave = threadIdx.x >> 5, lane = threadIdx.x & 31;
  int tn = blockIdx.x;
  int tm = blockIdx.y * 8 + wave;
  if (tm >= mtiles) return;
  const __bf16* pa = PA + ((size_t)tm * kchunks) * FRAG + lane * 16;
  const __bf16* pw = PW + (size_t)tn * FRAG + lane * 16;
  v8f c = {};
  for (int kc = 0; kc < kchunks; ++kc) {
    v16bf a = *(const v16bf*)(pa + (size_t)kc * FRAG);
    v16bf b = *(const v16bf*)(pw + (size_t)kc * ntiles * FRAG);
    c = __builtin_amdgcn_wmma_f32_16x16x32_bf16(false, a, false, b,
                                                (short)0, c, false, false);
  }
  int n = lane & 15, g = (lane >> 4) & 1;
#pragma unroll
  for (int r = 0; r < 8; ++r)
    Y[(size_t)(tm * 16 + 8 * g + r) * Nout + tn * 16 + n] = c[r];
}

// ---------------------------------------------------------------------------
// GCN support kernels
// ---------------------------------------------------------------------------
__global__ void edge_deg(const int* __restrict__ dst, float* __restrict__ deg,
                         int E) {
  int e = blockIdx.x * blockDim.x + threadIdx.x;
  if (e < E) atomicAdd(&deg[dst[e]], 1.0f);
}

__global__ void rsqrt_inplace(float* __restrict__ d, int n) {
  int i = blockIdx.x * blockDim.x + threadIdx.x;
  if (i < n) d[i] = rsqrtf(d[i]);
}

// AGG[dst] += H[src] * dinv[src]*dinv[dst]; one thread per (edge, 4 channels).
__global__ void gcn_aggregate4(const float* __restrict__ H,
                               float* __restrict__ AGG,
                               const int* __restrict__ src,
                               const int* __restrict__ dst,
                               const float* __restrict__ dinv, int E, int F) {
  int i = blockIdx.x * blockDim.x + threadIdx.x;
  int fg = F >> 2;
  int e = i / fg;
  if (e >= E) return;
  int f4 = (i - e * fg) << 2;
  int s = src[e], d = dst[e];
  float norm = dinv[s] * dinv[d];
  const float4 h = *(const float4*)(H + (size_t)s * F + f4);
  float* o = AGG + (size_t)d * F + f4;
  atomicAdd(o + 0, h.x * norm);
  atomicAdd(o + 1, h.y * norm);
  atomicAdd(o + 2, h.z * norm);
  atomicAdd(o + 3, h.w * norm);
}

// OUT = (AGG + H*dinv^2 + b) with optional ReLU
__global__ void gcn_combine(const float* __restrict__ AGG,
                            const float* __restrict__ H,
                            const float* __restrict__ dinv,
                            const float* __restrict__ bias,
                            float* __restrict__ OUT, int Nn, int F, int relu) {
  int i = blockIdx.x * blockDim.x + threadIdx.x;
  if (i >= Nn * F) return;
  int node = i / F, f = i - node * F;
  float dv = dinv[node];
  float v = AGG[i] + H[i] * dv * dv + bias[f];
  if (relu) v = fmaxf(v, 0.f);
  OUT[i] = v;
}

// ---------------------------------------------------------------------------
// Attention pooling: ctx = tanh(colmean(a) @ W); pooled = a^T @ sigmoid(a@ctx)
// ---------------------------------------------------------------------------
__global__ void colsum32(const float* __restrict__ A, float* __restrict__ out,
                         int Nn) {
  __shared__ float ls[32];
  if (threadIdx.x < 32) ls[threadIdx.x] = 0.f;
  __syncthreads();
  int i = blockIdx.x * blockDim.x + threadIdx.x;
  if (i < Nn * 32) atomicAdd(&ls[i & 31], A[i]);
  __syncthreads();
  if (threadIdx.x < 32) atomicAdd(&out[threadIdx.x], ls[threadIdx.x]);
}

__global__ void ctx_kernel(const float* __restrict__ colsum,
                           const float* __restrict__ W, float ninv,
                           float* __restrict__ ctx) {
  int f = threadIdx.x;  // 32 threads
  float t = 0.f;
  for (int i = 0; i < 32; ++i) t += colsum[i] * ninv * W[i * 32 + f];
  ctx[f] = tanhf(t);
}

__global__ void pool_kernel(const float* __restrict__ A,
                            const float* __restrict__ ctx,
                            float* __restrict__ P, int Nn) {
  __shared__ float lp[32];
  if (threadIdx.x < 32) lp[threadIdx.x] = 0.f;
  __syncthreads();
  float acc[32];
#pragma unroll
  for (int f = 0; f < 32; ++f) acc[f] = 0.f;
  for (int row = blockIdx.x * blockDim.x + threadIdx.x; row < Nn;
       row += gridDim.x * blockDim.x) {
    const float* r = A + (size_t)row * 32;
    float s = 0.f;
#pragma unroll
    for (int f = 0; f < 32; ++f) s += r[f] * ctx[f];
    float sg = 1.f / (1.f + expf(-s));
#pragma unroll
    for (int f = 0; f < 32; ++f) acc[f] += r[f] * sg;
  }
#pragma unroll
  for (int f = 0; f < 32; ++f) atomicAdd(&lp[f], acc[f]);
  __syncthreads();
  if (threadIdx.x < 32) atomicAdd(&P[threadIdx.x], lp[threadIdx.x]);
}

// ---------------------------------------------------------------------------
// Similarity-score passes on PACKED fragments.  Each wave owns MT=4 resident
// A fragments and streams B tiles: one 32B/lane B load -> 4 WMMAs.  Binning /
// min-max VALU co-executes with the 16-bit WMMA (TRANS-class) pipe.
// ---------------------------------------------------------------------------
__global__ void score_minmax(const __bf16* __restrict__ PA,
                             const __bf16* __restrict__ PB,
                             unsigned* __restrict__ mm, int mtiles, int ntiles,
                             int chunk) {
  int wave = threadIdx.x >> 5, lane = threadIdx.x & 31;
  int tmb = (blockIdx.y * 8 + wave) * MT;
  if (tmb >= mtiles) return;
  int nv = min(MT, mtiles - tmb);
  int n0 = blockIdx.x * chunk;
  int n1 = min(n0 + chunk, ntiles);
  v16bf a[MT];
#pragma unroll
  for (int t = 0; t < MT; ++t)
    a[t] = *(const v16bf*)(PA + (size_t)min(tmb + t, mtiles - 1) * FRAG +
                           lane * 16);
  const __bf16* pb = PB + (size_t)n0 * FRAG + lane * 16;
  float mn = 3.4e38f, mx = -3.4e38f;
  for (int tn = n0; tn < n1; ++tn, pb += FRAG) {
    if (tn + 1 < n1) __builtin_prefetch(pb + FRAG, 0, 0);
    v16bf b = *(const v16bf*)pb;
#pragma unroll
    for (int t = 0; t < MT; ++t) {
      if (t < nv) {
        v8f c = {};
        c = __builtin_amdgcn_wmma_f32_16x16x32_bf16(false, a[t], false, b,
                                                    (short)0, c, false, false);
#pragma unroll
        for (int r = 0; r < 8; ++r) {
          mn = fminf(mn, c[r]);
          mx = fmaxf(mx, c[r]);
        }
      }
    }
  }
  for (int off = 16; off; off >>= 1) {
    mn = fminf(mn, __shfl_xor(mn, off, 32));
    mx = fmaxf(mx, __shfl_xor(mx, off, 32));
  }
  if (lane == 0) {
    atomicMin(&mm[0], fkey(mn));
    atomicMax(&mm[1], fkey(mx));
  }
}

__global__ void score_hist(const __bf16* __restrict__ PA,
                           const __bf16* __restrict__ PB,
                           const unsigned* __restrict__ mm,
                           float* __restrict__ hist, int mtiles, int ntiles,
                           int chunk) {
  __shared__ unsigned lh[8][16];
  int tid = threadIdx.x;
  if (tid < 128) lh[tid >> 4][tid & 15] = 0u;
  __syncthreads();
  int wave = tid >> 5, lane = tid & 31;
  int tmb = (blockIdx.y * 8 + wave) * MT;
  bool active = (tmb < mtiles);
  float lo = fromkey(mm[0]);
  float hi = fromkey(mm[1]);
  float scale = 16.0f / fmaxf(hi - lo, 1e-12f);
  float base = -lo * scale;  // idx = floor(c*scale + base)
  if (active) {
    int nv = min(MT, mtiles - tmb);
    int n0 = blockIdx.x * chunk;
    int n1 = min(n0 + chunk, ntiles);
    v16bf a[MT];
#pragma unroll
    for (int t = 0; t < MT; ++t)
      a[t] = *(const v16bf*)(PA + (size_t)min(tmb + t, mtiles - 1) * FRAG +
                             lane * 16);
    const __bf16* pb = PB + (size_t)n0 * FRAG + lane * 16;
    for (int tn = n0; tn < n1; ++tn, pb += FRAG) {
      if (tn + 1 < n1) __builtin_prefetch(pb + FRAG, 0, 0);
      v16bf b = *(const v16bf*)pb;
#pragma unroll
      for (int t = 0; t < MT; ++t) {
        if (t < nv) {
          v8f c = {};
          c = __builtin_amdgcn_wmma_f32_16x16x32_bf16(
              false, a[t], false, b, (short)0, c, false, false);
#pragma unroll
          for (int r = 0; r < 8; ++r) {
            int idx = (int)floorf(__builtin_fmaf(c[r], scale, base));
            idx = min(max(idx, 0), 15);
            atomicAdd(&lh[wave][idx], 1u);
          }
        }
      }
    }
  }
  __syncthreads();
  if (tid < 16) {
    unsigned t = 0;
#pragma unroll
    for (int w = 0; w < 8; ++w) t += lh[w][tid];
    atomicAdd(&hist[tid], (float)t);
  }
}

// ---------------------------------------------------------------------------
// NTN + histogram normalize + MLP head (tiny; 1 block of 32 threads)
// ---------------------------------------------------------------------------
__global__ void final_kernel(const float* __restrict__ P1,
                             const float* __restrict__ P2,
                             const float* __restrict__ hist,
                             const float* __restrict__ ntn_w,
                             const float* __restrict__ ntn_v,
                             const float* __restrict__ ntn_b,
                             const float* __restrict__ fc1_w,
                             const float* __restrict__ fc1_b,
                             const float* __restrict__ fc2_w,
                             const float* __restrict__ fc2_b,
                             const float* __restrict__ fc3_w,
                             const float* __restrict__ fc3_b,
                             const float* __restrict__ score_w,
                             const float* __restrict__ score_b,
                             const float* __restrict__ avg_v,
                             float* __restrict__ out) {
  __shared__ float s[32];
  __shared__ float hs;
  int t = threadIdx.x;
  if (t < 16) {
    float acc = ntn_b[t];
    for (int i = 0; i < 32; ++i) {
      float p1i = P1[i];
      for (int j = 0; j < 32; ++j)
        acc += p1i * ntn_w[(i * 32 + j) * 16 + t] * P2[j];
    }
    for (int i = 0; i < 32; ++i) acc += ntn_v[t * 64 + i] * P1[i];
    for (int j = 0; j < 32; ++j) acc += ntn_v[t * 64 + 32 + j] * P2[j];
    s[t] = fmaxf(acc, 0.f);
  }
  if (t == 0) {
    float tot = 0.f;
    for (int b = 0; b < 16; ++b) tot += hist[b];
    hs = tot;
  }
  __syncthreads();
  if (t < 16) s[16 + t] = hist[t] / hs;
  __syncthreads();
  if (t == 0) {
    float x1[16], x2[8], x3[4];
    for (int o = 0; o < 16; ++o) {
      float a = fc1_b[o];
      for (int i = 0; i < 32; ++i) a += s[i] * fc1_w[i * 16 + o];
      x1[o] = fmaxf(a, 0.f);
    }
    for (int o = 0; o < 8; ++o) {
      float a = fc2_b[o];
      for (int i = 0; i < 16; ++i) a += x1[i] * fc2_w[i * 8 + o];
      x2[o] = fmaxf(a, 0.f);
    }
    for (int o = 0; o < 4; ++o) {
      float a = fc3_b[o];
      for (int i = 0; i < 8; ++i) a += x2[i] * fc3_w[i * 4 + o];
      x3[o] = fmaxf(a, 0.f);
    }
    float sc = score_b[0];
    for (int i = 0; i < 4; ++i) sc += x3[i] * score_w[i];
    sc = 1.f / (1.f + expf(-sc));
    out[0] = sc;
    out[1] = -logf(sc) * avg_v[0];
  }
}

// ---------------------------------------------------------------------------
// Host-side orchestration
// ---------------------------------------------------------------------------
static inline void fill_f(float* p, float v, long n, hipStream_t s) {
  fill_u32<<<dim3((unsigned)((n + 255) / 256)), 256, 0, s>>>(
      (unsigned*)p, __builtin_bit_cast(unsigned, v), n);
}

// one dense layer: pack X + W fragments, then WMMA GEMM
static void run_gemm(const float* X, const float* W, float* Y, int M, int K,
                     int Nout, __bf16* PA, __bf16* PW, hipStream_t s) {
  int mtiles = M / 16, kchunks = K / 32, ntiles = Nout / 16;
  int afrags = mtiles * kchunks, bfrags = kchunks * ntiles;
  pack_a_frags<<<(afrags + 7) / 8, 256, 0, s>>>(X, PA, mtiles, K);
  pack_b_frags<<<(bfrags + 7) / 8, 256, 0, s>>>(W, PW, K, Nout);
  gemm_packed<<<dim3((unsigned)ntiles, (unsigned)((mtiles + 7) / 8)), 256, 0,
                s>>>(PA, PW, Y, mtiles, kchunks, ntiles, Nout);
}

static void run_conv(const float* feat, const int* ei, int N, int E,
                     const float* w1, const float* b1, const float* w2,
                     const float* b2, const float* w3, const float* b3,
                     float* H, float* AGG, float* dinv, float* ACT,
                     float* out32, __bf16* PA, __bf16* PW, hipStream_t s) {
  const int* src = ei;
  const int* dst = ei + E;
  // degrees (with self loop) -> dinv
  fill_f(dinv, 1.0f, N, s);
  edge_deg<<<(E + 255) / 256, 256, 0, s>>>(dst, dinv, E);
  rsqrt_inplace<<<(N + 255) / 256, 256, 0, s>>>(dinv, N);

  // layer 1: 128 -> 128, relu
  run_gemm(feat, w1, H, N, 128, 128, PA, PW, s);
  fill_f(AGG, 0.f, (long)N * 128, s);
  gcn_aggregate4<<<(int)(((long)E * 32 + 255) / 256), 256, 0, s>>>(
      H, AGG, src, dst, dinv, E, 128);
  gcn_combine<<<(int)(((long)N * 128 + 255) / 256), 256, 0, s>>>(
      AGG, H, dinv, b1, ACT, N, 128, 1);

  // layer 2: 128 -> 64, relu
  run_gemm(ACT, w2, H, N, 128, 64, PA, PW, s);
  fill_f(AGG, 0.f, (long)N * 64, s);
  gcn_aggregate4<<<(int)(((long)E * 16 + 255) / 256), 256, 0, s>>>(
      H, AGG, src, dst, dinv, E, 64);
  gcn_combine<<<(int)(((long)N * 64 + 255) / 256), 256, 0, s>>>(
      AGG, H, dinv, b2, ACT, N, 64, 1);

  // layer 3: 64 -> 32, no relu
  run_gemm(ACT, w3, H, N, 64, 32, PA, PW, s);
  fill_f(AGG, 0.f, (long)N * 32, s);
  gcn_aggregate4<<<(int)(((long)E * 8 + 255) / 256), 256, 0, s>>>(
      H, AGG, src, dst, dinv, E, 32);
  gcn_combine<<<(int)(((long)N * 32 + 255) / 256), 256, 0, s>>>(
      AGG, H, dinv, b3, out32, N, 32, 0);
}

static void run_pool(const float* A, const float* attn_w, int N, float* colsum,
                     float* ctx, float* P, hipStream_t s) {
  fill_f(colsum, 0.f, 32, s);
  colsum32<<<(int)(((long)N * 32 + 255) / 256), 256, 0, s>>>(A, colsum, N);
  ctx_kernel<<<1, 32, 0, s>>>(colsum, attn_w, 1.0f / (float)N, ctx);
  fill_f(P, 0.f, 32, s);
  pool_kernel<<<256, 256, 0, s>>>(A, ctx, P, N);
}

extern "C" void kernel_launch(void* const* d_in, const int* in_sizes, int n_in,
                              void* d_out, int out_size, void* d_ws,
                              size_t ws_size, hipStream_t stream) {
  const float* f1      = (const float*)d_in[0];
  const float* f2      = (const float*)d_in[1];
  const int*   ei1     = (const int*)d_in[2];
  const int*   ei2     = (const int*)d_in[3];
  const float* avg_v   = (const float*)d_in[4];
  const float* w1      = (const float*)d_in[5];
  const float* b1      = (const float*)d_in[6];
  const float* w2      = (const float*)d_in[7];
  const float* b2      = (const float*)d_in[8];
  const float* w3      = (const float*)d_in[9];
  const float* b3      = (const float*)d_in[10];
  const float* attn_w  = (const float*)d_in[11];
  const float* ntn_w   = (const float*)d_in[12];
  const float* ntn_v   = (const float*)d_in[13];
  const float* ntn_b   = (const float*)d_in[14];
  const float* fc1_w   = (const float*)d_in[15];
  const float* fc1_b   = (const float*)d_in[16];
  const float* fc2_w   = (const float*)d_in[17];
  const float* fc2_b   = (const float*)d_in[18];
  const float* fc3_w   = (const float*)d_in[19];
  const float* fc3_b   = (const float*)d_in[20];
  const float* score_w = (const float*)d_in[21];
  const float* score_b = (const float*)d_in[22];

  int N = in_sizes[0] / 128;  // 20000 (multiple of 16)
  int E = in_sizes[2] / 2;    // 640000
  int mtiles = N / 16;        // 1250

  // workspace carve-up (fp32 region first, then 32B-aligned bf16 region)
  float* ws     = (float*)d_ws;
  float* H      = ws;                     // N*128
  float* AGG    = H + (size_t)N * 128;    // N*128
  float* ACT    = AGG + (size_t)N * 128;  // N*128
  float* A1     = ACT + (size_t)N * 128;  // N*32
  float* A2     = A1 + (size_t)N * 32;    // N*32
  float* dinv   = A2 + (size_t)N * 32;    // N
  float* colsum = dinv + N;               // 32
  float* ctx    = colsum + 32;            // 32
  float* P1     = ctx + 32;               // 32
  float* P2     = P1 + 32;                // 32
  float* hist   = P2 + 32;                // 16
  unsigned* mm  = (unsigned*)(hist + 16); // 2 (+ pad to 32B below)
  size_t fcount = (size_t)(hist + 16 + 2 + 8 - (float*)d_ws);
  fcount = (fcount + 7) & ~(size_t)7;     // 32B align for v16bf loads
  __bf16* PA  = (__bf16*)((float*)d_ws + fcount);       // N/16*K/32 frags (GEMM X)
  __bf16* PW  = PA + (size_t)mtiles * 4 * FRAG;         // up to 32 frags (GEMM W)
  __bf16* PH1 = PW + (size_t)32 * FRAG;                 // mtiles frags (a1)
  __bf16* PH2 = PH1 + (size_t)mtiles * FRAG;            // mtiles frags (a2^T)

  // conv towers (shared weights)
  run_conv(f1, ei1, N, E, w1, b1, w2, b2, w3, b3, H, AGG, dinv, ACT, A1, PA,
           PW, stream);
  run_conv(f2, ei2, N, E, w1, b1, w2, b2, w3, b3, H, AGG, dinv, ACT, A2, PA,
           PW, stream);

  // attention pooling
  run_pool(A1, attn_w, N, colsum, ctx, P1, stream);
  run_pool(A2, attn_w, N, colsum, ctx, P2, stream);

  // histogram of a1 @ a2^T: pack fragments once, then two streaming passes
  pack_a_frags<<<(mtiles + 7) / 8, 256, 0, stream>>>(A1, PH1, mtiles, 32);
  pack_bt_frags<<<(mtiles + 7) / 8, 256, 0, stream>>>(A2, PH2, mtiles);
  fill_u32<<<1, 32, 0, stream>>>(mm, 0xFFFFFFFFu, 1);  // min key init
  fill_u32<<<1, 32, 0, stream>>>(mm + 1, 0u, 1);       // max key init
  fill_f(hist, 0.f, 16, stream);
  {
    const int CHUNK = 64;
    int nchunks = (mtiles + CHUNK - 1) / CHUNK;        // 20 for N=20000
    int mgroups = (mtiles + MT - 1) / MT;              // 313
    dim3 grid((unsigned)nchunks, (unsigned)((mgroups + 7) / 8));
    score_minmax<<<grid, 256, 0, stream>>>(PH1, PH2, mm, mtiles, mtiles, CHUNK);
    score_hist<<<grid, 256, 0, stream>>>(PH1, PH2, mm, hist, mtiles, mtiles,
                                         CHUNK);
  }

  // NTN + MLP head
  final_kernel<<<1, 32, 0, stream>>>(P1, P2, hist, ntn_w, ntn_v, ntn_b, fc1_w,
                                     fc1_b, fc2_w, fc2_b, fc3_w, fc3_b,
                                     score_w, score_b, avg_v, (float*)d_out);
}
